// SymptomRoutedRGAT_472446403321
// MI455X (gfx1250) — matrix operand order repaired and checked by reading, not compiled, not run on hardware
//
#include <hip/hip_runtime.h>
#include <hip/hip_bf16.h>
#include <math.h>

#define N_NODES 20000
#define HDIM    256
#define RREL    4
#define KSYM    8
#define NHEAD   4
#define HEADD   64
#define NEDGE   320000
#define INV_SCALE 0.125f   // 1/sqrt(64)
#define LN_EPS  1e-5f

typedef __attribute__((ext_vector_type(16))) __bf16 v16bf;
typedef __attribute__((ext_vector_type(8)))  __bf16 v8bf;
typedef __attribute__((ext_vector_type(8)))  float  v8f;

// ---------------------------------------------------------------- helpers
__device__ __forceinline__ __bf16 f2bf(float f) {
    unsigned u = __float_as_uint(f);
    unsigned r = u + 0x7FFFu + ((u >> 16) & 1u);   // round-to-nearest-even
    unsigned short hb = (unsigned short)(r >> 16);
    return __builtin_bit_cast(__bf16, hb);
}

// order-preserving float<->uint key for atomicMax over signed floats
__device__ __forceinline__ unsigned fkey(float f) {
    unsigned u = __float_as_uint(f);
    return (u & 0x80000000u) ? ~u : (u | 0x80000000u);
}
__device__ __forceinline__ float finv(unsigned u) {
    unsigned b = (u & 0x80000000u) ? (u & 0x7FFFFFFFu) : ~u;
    return __uint_as_float(b);
}

// ---------------------------------------------------------------- utility kernels
__global__ void zero_u32_kernel(unsigned* __restrict__ p, size_t n) {
    size_t i = (size_t)blockIdx.x * blockDim.x + threadIdx.x;
    if (i < n) p[i] = 0u;
}

__global__ void cvt_bf16_kernel(const float* __restrict__ in, __bf16* __restrict__ out, size_t n) {
    size_t i = (size_t)blockIdx.x * blockDim.x + threadIdx.x;
    if (i < n) out[i] = f2bf(in[i]);
}

// softmax over R for each of K rows of sym_edge_logits
__global__ void symw_kernel(const float* __restrict__ logits, float* __restrict__ symw) {
    int k = threadIdx.x;
    if (k >= KSYM) return;
    float m = -3.4e38f;
    for (int r = 0; r < RREL; r++) m = fmaxf(m, logits[k * RREL + r]);
    float e[RREL]; float s = 0.f;
    for (int r = 0; r < RREL; r++) { e[r] = __expf(logits[k * RREL + r] - m); s += e[r]; }
    float inv = 1.f / s;
    for (int r = 0; r < RREL; r++) symw[k * RREL + r] = e[r] * inv;
}

// ---------------------------------------------------------------- generic bf16 WMMA GEMM
// out(M x 256) = A(M x Kd, bf16 row-major) * W(256 x Kd, bf16 row-major)^T
// Per wave: 32 rows x 64 cols (2 A frags, 4 B tiles shared -> 8 WMMA per K-step of 32).
// mode 0: f32 store            (QKV projections)
// mode 1: +bias, relu, f32     (MLP hidden, cross linear)
// mode 2: +bias, relu, *gate(row), bf16 store  (symptom channels into cat)
__global__ __launch_bounds__(256)
void gemm_bf16_kernel(const __bf16* __restrict__ A,
                      const __bf16* __restrict__ W,
                      const float*  __restrict__ bias,
                      const float*  __restrict__ gatep, int gstride,
                      float*  __restrict__ outF,
                      __bf16* __restrict__ outB,
                      int M, int Kd, int ldc, int mode)
{
    const int lane = threadIdx.x & 31;
    const int wave = threadIdx.x >> 5;
    const int m0 = blockIdx.x * 256 + wave * 32;   // 32 rows per wave (32 | M)
    const int n0 = blockIdx.y * 64;
    if (m0 >= M) return;

    const int half = lane >> 4;                 // 0: lanes 0-15, 1: lanes 16-31
    const int l15  = lane & 15;

    // A fragment sources: rows m0+l15 and m0+16+l15; K halves offset by +16 for upper lanes
    const __bf16* arow0 = A + (size_t)(m0 + l15) * Kd + half * 16;
    const __bf16* arow1 = arow0 + (size_t)16 * Kd;
    // B fragment source: column n0+l15 -> row (n0+l15) of W; K offset +16 for upper lanes
    const __bf16* brow = W + (size_t)(n0 + l15) * Kd + half * 16;
    const size_t bstride16 = (size_t)16 * Kd;   // step 16 output columns

    v8f acc[2][4];
#pragma unroll
    for (int t = 0; t < 2; t++)
#pragma unroll
        for (int j = 0; j < 4; j++) acc[t][j] = (v8f){};

    for (int k0 = 0; k0 < Kd; k0 += 32) {
        // A 16x32 bf16 fragments (contiguous: e0..15 -> K=base..base+15, base = k0 + 16*half)
        v8bf a0lo = *(const v8bf*)(arow0 + k0);
        v8bf a0hi = *(const v8bf*)(arow0 + k0 + 8);
        v8bf a1lo = *(const v8bf*)(arow1 + k0);
        v8bf a1hi = *(const v8bf*)(arow1 + k0 + 8);
        v16bf a0, a1;
#pragma unroll
        for (int i = 0; i < 8; i++) {
            a0[i] = a0lo[i]; a0[i + 8] = a0hi[i];
            a1[i] = a1lo[i]; a1[i + 8] = a1hi[i];
        }

        // B 32x16 bf16 fragments: element e -> K = k0 + e (+16 upper lanes), contiguous 32B
        v16bf b[4];
#pragma unroll
        for (int j = 0; j < 4; j++)
            b[j] = *(const v16bf*)(brow + k0 + (size_t)j * bstride16);

#pragma unroll
        for (int j = 0; j < 4; j++) {
            acc[0][j] = __builtin_amdgcn_wmma_f32_16x16x32_bf16(false, a0, false, b[j], (short)0, acc[0][j], false, false);
            acc[1][j] = __builtin_amdgcn_wmma_f32_16x16x32_bf16(false, a1, false, b[j], (short)0, acc[1][j], false, false);
        }
    }

    // C/D layout: element i -> row m0 + 16*t + i + 8*half, col n0 + 16*j + l15
#pragma unroll
    for (int t = 0; t < 2; t++) {
        const int rbase = m0 + 16 * t + half * 8;
#pragma unroll
        for (int j = 0; j < 4; j++) {
            int c = n0 + 16 * j + l15;
            float b = (mode >= 1) ? bias[c] : 0.f;
#pragma unroll
            for (int i = 0; i < 8; i++) {
                int row = rbase + i;
                float v = acc[t][j][i];
                if (mode >= 1) { v += b; v = fmaxf(v, 0.f); }
                if (mode == 2) {
                    v *= gatep[(size_t)row * gstride];
                    outB[(size_t)row * ldc + c] = f2bf(v);
                } else {
                    outF[(size_t)row * ldc + c] = v;
                }
            }
        }
    }
}

// ---------------------------------------------------------------- edge attention
// pass 1: per-(edge,head) score + segment max (via order-preserving uint atomicMax)
__global__ __launch_bounds__(256)
void score_kernel(const int* __restrict__ ei, const int* __restrict__ et,
                  const float* __restrict__ Q, const float* __restrict__ Kp,
                  float* __restrict__ score, unsigned* __restrict__ smaxu)
{
    int idx = blockIdx.x * blockDim.x + threadIdx.x;
    if (idx >= NEDGE * NHEAD) return;
    int e = idx >> 2, hh = idx & 3;
    int s = ei[e], d = ei[NEDGE + e], t = et[e];
    const float4* q4 = (const float4*)(Q  + ((size_t)t * N_NODES + d) * HDIM + hh * HEADD);
    const float4* k4 = (const float4*)(Kp + ((size_t)t * N_NODES + s) * HDIM + hh * HEADD);
    float acc = 0.f;
#pragma unroll
    for (int i = 0; i < HEADD / 4; i++) {
        float4 a = q4[i], b = k4[i];
        acc += a.x * b.x + a.y * b.y + a.z * b.z + a.w * b.w;
    }
    float sc = acc * INV_SCALE;
    score[idx] = sc;
    atomicMax(&smaxu[((size_t)t * N_NODES + d) * NHEAD + hh], fkey(sc));
}

// pass 2: exp(score - max), accumulate denominator
__global__ __launch_bounds__(256)
void expsum_kernel(const int* __restrict__ ei, const int* __restrict__ et,
                   float* __restrict__ score, const unsigned* __restrict__ smaxu,
                   float* __restrict__ denom)
{
    int idx = blockIdx.x * blockDim.x + threadIdx.x;
    if (idx >= NEDGE * NHEAD) return;
    int e = idx >> 2, hh = idx & 3;
    int d = ei[NEDGE + e], t = et[e];
    size_t seg = ((size_t)t * N_NODES + d) * NHEAD + hh;
    float ex = __expf(score[idx] - finv(smaxu[seg]));
    score[idx] = ex;
    atomicAdd(&denom[seg], ex);
}

// pass 3: alpha * V scatter-add into agg (R*N, H)
__global__ __launch_bounds__(256)
void aggregate_kernel(const int* __restrict__ ei, const int* __restrict__ et,
                      const float* __restrict__ score, const float* __restrict__ denom,
                      const float* __restrict__ V, float* __restrict__ agg)
{
    int idx = blockIdx.x * blockDim.x + threadIdx.x;
    if (idx >= NEDGE * NHEAD) return;
    int e = idx >> 2, hh = idx & 3;
    int s = ei[e], d = ei[NEDGE + e], t = et[e];
    size_t seg = ((size_t)t * N_NODES + d) * NHEAD + hh;
    float al = score[idx] / denom[seg];
    const float* v = V   + ((size_t)t * N_NODES + s) * HDIM + hh * HEADD;
    float*       a = agg + ((size_t)t * N_NODES + d) * HDIM + hh * HEADD;
#pragma unroll 8
    for (int i = 0; i < HEADD; i++) atomicAdd(&a[i], al * v[i]);
}

// mix over relations: mixbf[k][n][h] = sum_r symw[k][r] * agg[r][n][h]  (bf16)
__global__ __launch_bounds__(256)
void mix_kernel(const float* __restrict__ agg, const float* __restrict__ symw,
                __bf16* __restrict__ mixbf)
{
    size_t idx = (size_t)blockIdx.x * blockDim.x + threadIdx.x;
    const size_t NH_ = (size_t)N_NODES * HDIM;
    if (idx >= NH_) return;
    float a[RREL];
#pragma unroll
    for (int r = 0; r < RREL; r++) a[r] = agg[(size_t)r * NH_ + idx];
#pragma unroll
    for (int k = 0; k < KSYM; k++) {
        float v = 0.f;
#pragma unroll
        for (int r = 0; r < RREL; r++) v += symw[k * RREL + r] * a[r];
        mixbf[(size_t)k * NH_ + idx] = f2bf(v);
    }
}

// routing gate: softmax_k( hid @ mlp_w2.T + b2 ), one thread per node
__global__ __launch_bounds__(256)
void gate_kernel(const float* __restrict__ hid, const float* __restrict__ w2,
                 const float* __restrict__ b2, float* __restrict__ gate)
{
    int n = blockIdx.x * blockDim.x + threadIdx.x;
    if (n >= N_NODES) return;
    float lg[KSYM];
#pragma unroll
    for (int k = 0; k < KSYM; k++) lg[k] = b2[k];
    const float* hr = hid + (size_t)n * HDIM;
    for (int hh = 0; hh < HDIM; hh++) {
        float x = hr[hh];
#pragma unroll
        for (int k = 0; k < KSYM; k++) lg[k] += x * w2[k * HDIM + hh];
    }
    float m = lg[0];
#pragma unroll
    for (int k = 1; k < KSYM; k++) m = fmaxf(m, lg[k]);
    float s = 0.f;
#pragma unroll
    for (int k = 0; k < KSYM; k++) { lg[k] = __expf(lg[k] - m); s += lg[k]; }
    float inv = 1.f / s;
#pragma unroll
    for (int k = 0; k < KSYM; k++) gate[(size_t)n * KSYM + k] = lg[k] * inv;
}

// residual + LayerNorm, one wave32 per node
__global__ __launch_bounds__(256)
void ln_kernel(const float* __restrict__ h, const float* __restrict__ hnew,
               const float* __restrict__ gamma, const float* __restrict__ beta,
               float* __restrict__ out)
{
    int lane = threadIdx.x & 31;
    int wave = threadIdx.x >> 5;
    int n = blockIdx.x * 8 + wave;
    if (n >= N_NODES) return;
    const float* x1 = h    + (size_t)n * HDIM;
    const float* x2 = hnew + (size_t)n * HDIM;
    float xr[8], s = 0.f, ss = 0.f;
#pragma unroll
    for (int i = 0; i < 8; i++) {
        int hh = lane + 32 * i;
        float x = x1[hh] + x2[hh];
        xr[i] = x; s += x; ss += x * x;
    }
#pragma unroll
    for (int off = 16; off > 0; off >>= 1) {
        s  += __shfl_xor(s,  off, 32);
        ss += __shfl_xor(ss, off, 32);
    }
    float mu  = s * (1.f / HDIM);
    float var = ss * (1.f / HDIM) - mu * mu;
    float inv = rsqrtf(var + LN_EPS);
#pragma unroll
    for (int i = 0; i < 8; i++) {
        int hh = lane + 32 * i;
        out[(size_t)n * HDIM + hh] = (xr[i] - mu) * inv * gamma[hh] + beta[hh];
    }
}

// ---------------------------------------------------------------- launch
extern "C" void kernel_launch(void* const* d_in, const int* in_sizes, int n_in,
                              void* d_out, int out_size, void* d_ws, size_t ws_size,
                              hipStream_t stream) {
    (void)in_sizes; (void)n_in; (void)out_size; (void)ws_size;

    const float* h          = (const float*)d_in[0];
    const int*   edge_index = (const int*)  d_in[1];
    const int*   edge_type  = (const int*)  d_in[2];
    const float* W_q        = (const float*)d_in[3];
    const float* W_k        = (const float*)d_in[4];
    const float* W_v        = (const float*)d_in[5];
    const float* mlp_w1     = (const float*)d_in[6];
    const float* mlp_b1     = (const float*)d_in[7];
    const float* mlp_w2     = (const float*)d_in[8];
    const float* mlp_b2     = (const float*)d_in[9];
    const float* sym_logits = (const float*)d_in[10];
    const float* W_sym      = (const float*)d_in[11];
    const float* b_sym      = (const float*)d_in[12];
    const float* W_cross    = (const float*)d_in[13];
    const float* b_cross    = (const float*)d_in[14];
    const float* ln_gamma   = (const float*)d_in[15];
    const float* ln_beta    = (const float*)d_in[16];
    float* out = (float*)d_out;

    // ---- workspace carve-out
    char* ws = (char*)d_ws;
    size_t off = 0;
    auto alloc = [&](size_t bytes) -> char* {
        char* p = ws + off;
        off += (bytes + 255) & ~(size_t)255;
        return p;
    };
    const size_t NHt = (size_t)N_NODES * HDIM;
    __bf16* hbf     = (__bf16*)alloc(NHt * 2);
    __bf16* wqbf    = (__bf16*)alloc((size_t)RREL * HDIM * HDIM * 2);
    __bf16* wkbf    = (__bf16*)alloc((size_t)RREL * HDIM * HDIM * 2);
    __bf16* wvbf    = (__bf16*)alloc((size_t)RREL * HDIM * HDIM * 2);
    __bf16* w1bf    = (__bf16*)alloc((size_t)HDIM * HDIM * 2);
    __bf16* wsymbf  = (__bf16*)alloc((size_t)KSYM * HDIM * HDIM * 2);
    __bf16* wxbf    = (__bf16*)alloc((size_t)HDIM * KSYM * HDIM * 2);
    float*  Qf      = (float*)alloc((size_t)RREL * NHt * 4);
    float*  Kf      = (float*)alloc((size_t)RREL * NHt * 4);
    float*  Vf      = (float*)alloc((size_t)RREL * NHt * 4);
    float*  score   = (float*)alloc((size_t)NEDGE * NHEAD * 4);
    unsigned* smaxu = (unsigned*)alloc((size_t)RREL * N_NODES * NHEAD * 4);
    float*  denom   = (float*)alloc((size_t)RREL * N_NODES * NHEAD * 4);
    float*  agg     = (float*)alloc((size_t)RREL * NHt * 4);
    float*  symw    = (float*)alloc((size_t)KSYM * RREL * 4);
    __bf16* mixbf   = (__bf16*)alloc((size_t)KSYM * NHt * 2);
    float*  hid     = (float*)alloc(NHt * 4);
    float*  gate    = (float*)alloc((size_t)N_NODES * KSYM * 4);
    __bf16* catbf   = (__bf16*)alloc((size_t)N_NODES * KSYM * HDIM * 2);
    float*  hnew    = (float*)alloc(NHt * 4);

    auto blocks = [](size_t n) { return dim3((unsigned)((n + 255) / 256)); };
    auto cvt = [&](const float* in, __bf16* o, size_t n) {
        cvt_bf16_kernel<<<blocks(n), 256, 0, stream>>>(in, o, n);
    };

    // ---- precision conversions
    cvt(h, hbf, NHt);
    cvt(W_q, wqbf, (size_t)RREL * HDIM * HDIM);
    cvt(W_k, wkbf, (size_t)RREL * HDIM * HDIM);
    cvt(W_v, wvbf, (size_t)RREL * HDIM * HDIM);
    cvt(mlp_w1, w1bf, (size_t)HDIM * HDIM);
    cvt(W_sym, wsymbf, (size_t)KSYM * HDIM * HDIM);
    cvt(W_cross, wxbf, (size_t)HDIM * KSYM * HDIM);

    // ---- zero the scatter targets
    zero_u32_kernel<<<blocks((size_t)RREL * NHt), 256, 0, stream>>>((unsigned*)agg, (size_t)RREL * NHt);
    zero_u32_kernel<<<blocks((size_t)RREL * N_NODES * NHEAD), 256, 0, stream>>>(smaxu, (size_t)RREL * N_NODES * NHEAD);
    zero_u32_kernel<<<blocks((size_t)RREL * N_NODES * NHEAD), 256, 0, stream>>>((unsigned*)denom, (size_t)RREL * N_NODES * NHEAD);

    symw_kernel<<<1, 32, 0, stream>>>(sym_logits, symw);

    // ---- QKV projections (WMMA bf16 GEMMs), per relation
    dim3 gg((N_NODES + 255) / 256, HDIM / 64);
    for (int r = 0; r < RREL; r++) {
        const size_t wo = (size_t)r * HDIM * HDIM;
        gemm_bf16_kernel<<<gg, 256, 0, stream>>>(hbf, wqbf + wo, nullptr, nullptr, 0,
                                                 Qf + (size_t)r * NHt, nullptr, N_NODES, HDIM, HDIM, 0);
        gemm_bf16_kernel<<<gg, 256, 0, stream>>>(hbf, wkbf + wo, nullptr, nullptr, 0,
                                                 Kf + (size_t)r * NHt, nullptr, N_NODES, HDIM, HDIM, 0);
        gemm_bf16_kernel<<<gg, 256, 0, stream>>>(hbf, wvbf + wo, nullptr, nullptr, 0,
                                                 Vf + (size_t)r * NHt, nullptr, N_NODES, HDIM, HDIM, 0);
    }

    // ---- edge attention (scatter softmax + aggregation)
    dim3 eb = blocks((size_t)NEDGE * NHEAD);
    score_kernel    <<<eb, 256, 0, stream>>>(edge_index, edge_type, Qf, Kf, score, smaxu);
    expsum_kernel   <<<eb, 256, 0, stream>>>(edge_index, edge_type, score, smaxu, denom);
    aggregate_kernel<<<eb, 256, 0, stream>>>(edge_index, edge_type, score, denom, Vf, agg);

    // ---- relation mixing into K symptom streams (bf16)
    mix_kernel<<<blocks(NHt), 256, 0, stream>>>(agg, symw, mixbf);

    // ---- routing MLP + gate
    gemm_bf16_kernel<<<gg, 256, 0, stream>>>(hbf, w1bf, mlp_b1, nullptr, 0,
                                             hid, nullptr, N_NODES, HDIM, HDIM, 1);
    gate_kernel<<<blocks(N_NODES), 256, 0, stream>>>(hid, mlp_w2, mlp_b2, gate);

    // ---- K symptom linears: relu(mix_k @ W_sym[k].T + b_sym[k]) * gate[:,k] -> cat[:, k*H:(k+1)*H]
    for (int k = 0; k < KSYM; k++) {
        gemm_bf16_kernel<<<gg, 256, 0, stream>>>(mixbf + (size_t)k * NHt,
                                                 wsymbf + (size_t)k * HDIM * HDIM,
                                                 b_sym + (size_t)k * HDIM,
                                                 gate + k, KSYM,
                                                 nullptr, catbf + (size_t)k * HDIM,
                                                 N_NODES, HDIM, KSYM * HDIM, 2);
    }

    // ---- cross linear: relu(cat @ W_cross.T + b_cross)
    gemm_bf16_kernel<<<gg, 256, 0, stream>>>(catbf, wxbf, b_cross, nullptr, 0,
                                             hnew, nullptr, N_NODES, KSYM * HDIM, HDIM, 1);

    // ---- residual + LayerNorm
    ln_kernel<<<dim3((N_NODES + 7) / 8), 256, 0, stream>>>(h, hnew, ln_gamma, ln_beta, out);
}
